// DiffAttention_89739046683276
// MI455X (gfx1250) — matrix operand order repaired and checked by reading, not compiled, not run on hardware
//
#include <hip/hip_runtime.h>
#include <hip/hip_bf16.h>

// ---------------------------------------------------------------------------
// DiffAttention for MI455X (gfx1250): all matmuls via v_wmma_f32_16x16x32_f16.
//   B=2, N=2048, DIM=1024, H=16, hd=32 (qk), 64 (v)
// ---------------------------------------------------------------------------

typedef __attribute__((ext_vector_type(16))) _Float16 v16h;
typedef __attribute__((ext_vector_type(8)))  _Float16 v8h;
typedef __attribute__((ext_vector_type(4)))  _Float16 v4h;
typedef __attribute__((ext_vector_type(8)))  float    v8f;

#define LAMBDA_INIT 0.78360576682256813f
#define ONE_MINUS_LI 0.21639423317743187f
#define QK_SCALE 0.17677669529663687f   // 32^-0.5
#define SEQ_N 2048
#define NDIM 1024

static __device__ __forceinline__ void wave_lds_fence() {
  // DS ops from one wave are serviced in order; just make sure the compiler
  // doesn't reorder and the stores have drained before cross-lane reads.
  asm volatile("s_wait_dscnt 0x0" ::: "memory");
}

// ---------------------------------------------------------------------------
// 16-lane butterfly reductions in pure VALU via DPP16 (no LDS round trips).
// Steps: xor1 (quad_perm 1,0,3,2), xor2 (quad_perm 2,3,0,1), then
// row_half_mirror (== xor4 once quads are uniform), row_mirror (== xor8 once
// 8-groups are uniform). Valid on wave32 where a DPP row is 16 lanes.
// ---------------------------------------------------------------------------
#define DPP_QUAD_XOR1 0xB1
#define DPP_QUAD_XOR2 0x4E
#define DPP_ROW_HALF_MIRROR 0x141
#define DPP_ROW_MIRROR 0x140

static __device__ __forceinline__ float dpp_bcast_f32(float x, int ctrl_taken);

template <int CTRL>
static __device__ __forceinline__ float dpp_mov(float x) {
  int xi = __builtin_bit_cast(int, x);
  int r = __builtin_amdgcn_update_dpp(0, xi, CTRL, 0xf, 0xf, true);
  return __builtin_bit_cast(float, r);
}

static __device__ __forceinline__ float dpp_max16(float x) {
  x = fmaxf(x, dpp_mov<DPP_QUAD_XOR1>(x));
  x = fmaxf(x, dpp_mov<DPP_QUAD_XOR2>(x));
  x = fmaxf(x, dpp_mov<DPP_ROW_HALF_MIRROR>(x));
  x = fmaxf(x, dpp_mov<DPP_ROW_MIRROR>(x));
  return x;
}

static __device__ __forceinline__ float dpp_sum16(float x) {
  x += dpp_mov<DPP_QUAD_XOR1>(x);
  x += dpp_mov<DPP_QUAD_XOR2>(x);
  x += dpp_mov<DPP_ROW_HALF_MIRROR>(x);
  x += dpp_mov<DPP_ROW_MIRROR>(x);
  return x;
}

// ---------------------------------------------------------------------------
// f32 -> f16 cast (vectorized)
// ---------------------------------------------------------------------------
__global__ __launch_bounds__(256) void cast_f32_to_f16(const float* __restrict__ in,
                                                       _Float16* __restrict__ out, int n4) {
  int i = blockIdx.x * blockDim.x + threadIdx.x;
  if (i < n4) {
    const float4 v = ((const float4*)in)[i];
    v4h h; h[0] = (_Float16)v.x; h[1] = (_Float16)v.y; h[2] = (_Float16)v.z; h[3] = (_Float16)v.w;
    ((v4h*)out)[i] = h;
  }
}

// ---------------------------------------------------------------------------
// lambda scalar: exp(sum(lq1*lk1)) - exp(sum(lq2*lk2)) + LAMBDA_INIT
// ---------------------------------------------------------------------------
__global__ void lambda_kernel(const float* lq1, const float* lk1,
                              const float* lq2, const float* lk2, float* lam) {
  int d = threadIdx.x;             // 32 threads == HEAD_DIM
  float a = lq1[d] * lk1[d];
  float b = lq2[d] * lk2[d];
  #pragma unroll
  for (int m = 16; m >= 1; m >>= 1) {
    a += __shfl_xor(a, m, 32);
    b += __shfl_xor(b, m, 32);
  }
  if (d == 0) *lam = __expf(a) - __expf(b) + LAMBDA_INIT;
}

// ---------------------------------------------------------------------------
// Tiled WMMA GEMM: C(MxN) = A(MxK) @ B(KxN), f16 in, f32 accumulate.
// Block tile 128x128, 8 waves (4x2), wave tile 32x64, K step 32.
// OUT_F16: store f16 (no bias) ; else store f32 + bias.
// ---------------------------------------------------------------------------
template <bool OUT_F16>
__global__ __launch_bounds__(256) void wmma_gemm(const _Float16* __restrict__ A,
                                                 const _Float16* __restrict__ B,
                                                 _Float16* __restrict__ Ch,
                                                 float* __restrict__ Cf,
                                                 const float* __restrict__ bias,
                                                 int M, int Nn, int K) {
  __shared__ __align__(16) _Float16 As[128 * 32];   // row-major MxK tile
  __shared__ __align__(16) _Float16 Bt[128 * 32];   // transposed: [n][k]

  const int t = threadIdx.x;
  const int lane = t & 31, wid = t >> 5;
  const int l16 = lane & 15, lh = lane >> 4;
  const int wm = wid & 3, wn = wid >> 2;            // 4x2 wave grid
  const int m0 = blockIdx.y * 128, n0 = blockIdx.x * 128;

  v8f acc[2][4] = {};

  for (int k0 = 0; k0 < K; k0 += 32) {
    __syncthreads();
    // Stage A: 128x32 halves = 512 x v8h, 2 per thread.
    #pragma unroll
    for (int i = 0; i < 2; ++i) {
      int seg = t + i * 256;
      int row = seg >> 2, c8 = (seg & 3) * 8;
      v8h v = *(const v8h*)(A + (size_t)(m0 + row) * K + k0 + c8);
      *(v8h*)(As + row * 32 + c8) = v;
    }
    // Stage B transposed: 32x128 halves, scalar-transpose into Bt[n][k].
    #pragma unroll
    for (int i = 0; i < 2; ++i) {
      int seg = t + i * 256;
      int kr = seg >> 4, c8 = (seg & 15) * 8;
      v8h v = *(const v8h*)(B + (size_t)(k0 + kr) * Nn + n0 + c8);
      #pragma unroll
      for (int j = 0; j < 8; ++j) Bt[(c8 + j) * 32 + kr] = v[j];
    }
    __syncthreads();

    v16h af[2], bf[4];
    #pragma unroll
    for (int mi = 0; mi < 2; ++mi) {
      int row = wm * 32 + mi * 16 + l16;
      int kb = lh * 8;
      v8h lo = *(const v8h*)(As + row * 32 + kb);
      v8h hi = *(const v8h*)(As + row * 32 + kb + 16);
      #pragma unroll
      for (int j = 0; j < 8; ++j) { af[mi][j] = lo[j]; af[mi][j + 8] = hi[j]; }
    }
    #pragma unroll
    for (int ni = 0; ni < 4; ++ni) {
      int col = wn * 64 + ni * 16 + l16;
      int kb = lh * 8;
      v8h lo = *(const v8h*)(Bt + col * 32 + kb);
      v8h hi = *(const v8h*)(Bt + col * 32 + kb + 16);
      #pragma unroll
      for (int j = 0; j < 8; ++j) { bf[ni][j] = lo[j]; bf[ni][j + 8] = hi[j]; }
    }
    #pragma unroll
    for (int mi = 0; mi < 2; ++mi)
      #pragma unroll
      for (int ni = 0; ni < 4; ++ni)
        acc[mi][ni] = __builtin_amdgcn_wmma_f32_16x16x32_f16(
            false, af[mi], false, bf[ni], (short)0, acc[mi][ni], false, false);
  }

  // Epilogue (C layout: lane holds column l16, rows i + 8*lh per tile)
  #pragma unroll
  for (int mi = 0; mi < 2; ++mi)
    #pragma unroll
    for (int ni = 0; ni < 4; ++ni)
      #pragma unroll
      for (int i = 0; i < 8; ++i) {
        int row = m0 + wm * 32 + mi * 16 + 8 * lh + i;
        int col = n0 + wn * 64 + ni * 16 + l16;
        if (OUT_F16) Ch[(size_t)row * Nn + col] = (_Float16)acc[mi][ni][i];
        else         Cf[(size_t)row * Nn + col] = acc[mi][ni][i] + bias[col];
      }
}

// ---------------------------------------------------------------------------
// Flash differential attention.
// grid = (N/128, H, B); block = 256 (8 waves x 16 queries).
// qkv layout (f16): [b][n][3][H][64]  (3072 halves per token)
// NOTE: QK_SCALE is pre-folded into the Q fragments, so scores need no scale.
// ---------------------------------------------------------------------------
__device__ __forceinline__ void softmax_chunk(v8f sc[4], float mrun[8], float lsum[8],
                                              v8f oacc[4], _Float16* Pw, int l16, int lh) {
  #pragma unroll
  for (int i = 0; i < 8; ++i) {
    float mx = fmaxf(fmaxf(sc[0][i], sc[1][i]), fmaxf(sc[2][i], sc[3][i]));
    mx = dpp_max16(mx);                     // row max across the 16 key columns
    float mn = fmaxf(mrun[i], mx);
    float alpha = __expf(mrun[i] - mn);
    mrun[i] = mn;
    lsum[i] *= alpha;
    #pragma unroll
    for (int nt = 0; nt < 4; ++nt) oacc[nt][i] *= alpha;
  }
  #pragma unroll
  for (int kt = 0; kt < 4; ++kt)
    #pragma unroll
    for (int i = 0; i < 8; ++i) {
      float p = __expf(sc[kt][i] - mrun[i]);
      lsum[i] += p;
      Pw[(i + 8 * lh) * 64 + kt * 16 + l16] = (_Float16)p;
    }
}

__global__ __launch_bounds__(256) void diff_attn_kernel(const _Float16* __restrict__ qkv,
                                                        const float* __restrict__ lamp,
                                                        const float* __restrict__ subw,
                                                        _Float16* __restrict__ outh) {
  __shared__ __align__(16) _Float16 Ks1[64 * 32];      // [key][d]
  __shared__ __align__(16) _Float16 Ks2[64 * 32];
  __shared__ __align__(16) _Float16 Vt[64 * 64];       // [vdim][key]
  __shared__ __align__(16) _Float16 P1[8][16 * 64];    // per-wave P scratch
  __shared__ __align__(16) _Float16 P2[8][16 * 64];

  const int t = threadIdx.x;
  const int lane = t & 31, wid = t >> 5;
  const int l16 = lane & 15, lh = lane >> 4;
  const int b = blockIdx.z, h = blockIdx.y, q0 = blockIdx.x * 128;
  const int qw = q0 + wid * 16;

  // per-token half offsets inside qkv row (3072 halves)
  const int q1off = (h >> 1) * 64 + (h & 1) * 32;
  const int q2off = (8 + (h >> 1)) * 64 + (h & 1) * 32;
  const int k1off = 1024 + q1off;
  const int k2off = 1024 + q2off;
  const int voff  = 2048 + h * 64;

  // Q fragments (A layout), loaded once; fold in the softmax scale.
  v16h q1f, q2f;
  {
    const size_t qb = ((size_t)(b * SEQ_N + qw + l16)) * 3072;
    const int kb = lh * 8;
    v8h lo = *(const v8h*)(qkv + qb + q1off + kb);
    v8h hi = *(const v8h*)(qkv + qb + q1off + kb + 16);
    #pragma unroll
    for (int j = 0; j < 8; ++j) {
      q1f[j]     = (_Float16)((float)lo[j] * QK_SCALE);
      q1f[j + 8] = (_Float16)((float)hi[j] * QK_SCALE);
    }
    lo = *(const v8h*)(qkv + qb + q2off + kb);
    hi = *(const v8h*)(qkv + qb + q2off + kb + 16);
    #pragma unroll
    for (int j = 0; j < 8; ++j) {
      q2f[j]     = (_Float16)((float)lo[j] * QK_SCALE);
      q2f[j + 8] = (_Float16)((float)hi[j] * QK_SCALE);
    }
  }

  float m1[8], m2[8], s1[8], s2[8];
  v8f o1[4] = {}, o2[4] = {};
  #pragma unroll
  for (int i = 0; i < 8; ++i) { m1[i] = -1e30f; m2[i] = -1e30f; s1[i] = 0.f; s2[i] = 0.f; }

  const float lam = *lamp;

  for (int c0 = 0; c0 < SEQ_N; c0 += 64) {
    __syncthreads();
    // ---- stage K1/K2 (64x32 each): 1 v8h per thread per buffer
    {
      int kk = t >> 2, d8 = (t & 3) * 8;
      size_t kbase = ((size_t)(b * SEQ_N + c0 + kk)) * 3072;
      *(v8h*)(Ks1 + kk * 32 + d8) = *(const v8h*)(qkv + kbase + k1off + d8);
      *(v8h*)(Ks2 + kk * 32 + d8) = *(const v8h*)(qkv + kbase + k2off + d8);
      if (c0 + 64 < SEQ_N)  // warm next chunk
        __builtin_prefetch(qkv + kbase + 64 * 3072 + k1off, 0, 1);
    }
    // ---- stage V transposed (64 keys x 64 dims -> Vt[d][key])
    #pragma unroll
    for (int i = 0; i < 2; ++i) {
      int seg = t + i * 256;
      int kk = seg >> 3, d8 = (seg & 7) * 8;
      v8h v = *(const v8h*)(qkv + ((size_t)(b * SEQ_N + c0 + kk)) * 3072 + voff + d8);
      #pragma unroll
      for (int j = 0; j < 8; ++j) Vt[(d8 + j) * 64 + kk] = v[j];
    }
    __syncthreads();

    // ---- scores branch 1 (16q x 64k = 4 WMMAs), then softmax
    {
      v8f sc[4];
      #pragma unroll
      for (int kt = 0; kt < 4; ++kt) {
        int key = kt * 16 + l16, kb = lh * 8;
        v8h lo = *(const v8h*)(Ks1 + key * 32 + kb);
        v8h hi = *(const v8h*)(Ks1 + key * 32 + kb + 16);
        v16h bf;
        #pragma unroll
        for (int j = 0; j < 8; ++j) { bf[j] = lo[j]; bf[j + 8] = hi[j]; }
        v8f z = {};
        sc[kt] = __builtin_amdgcn_wmma_f32_16x16x32_f16(false, q1f, false, bf, (short)0, z, false, false);
      }
      softmax_chunk(sc, m1, s1, o1, P1[wid], l16, lh);
    }
    // ---- scores branch 2
    {
      v8f sc[4];
      #pragma unroll
      for (int kt = 0; kt < 4; ++kt) {
        int key = kt * 16 + l16, kb = lh * 8;
        v8h lo = *(const v8h*)(Ks2 + key * 32 + kb);
        v8h hi = *(const v8h*)(Ks2 + key * 32 + kb + 16);
        v16h bf;
        #pragma unroll
        for (int j = 0; j < 8; ++j) { bf[j] = lo[j]; bf[j + 8] = hi[j]; }
        v8f z = {};
        sc[kt] = __builtin_amdgcn_wmma_f32_16x16x32_f16(false, q2f, false, bf, (short)0, z, false, false);
      }
      softmax_chunk(sc, m2, s2, o2, P2[wid], l16, lh);
    }
    wave_lds_fence();

    // ---- P fragments (A layout) from per-wave LDS scratch
    v16h p1f[2], p2f[2];
    #pragma unroll
    for (int kf = 0; kf < 2; ++kf) {
      int base = kf * 32 + lh * 8;
      v8h lo = *(const v8h*)(P1[wid] + l16 * 64 + base);
      v8h hi = *(const v8h*)(P1[wid] + l16 * 64 + base + 16);
      #pragma unroll
      for (int j = 0; j < 8; ++j) { p1f[kf][j] = lo[j]; p1f[kf][j + 8] = hi[j]; }
      lo = *(const v8h*)(P2[wid] + l16 * 64 + base);
      hi = *(const v8h*)(P2[wid] + l16 * 64 + base + 16);
      #pragma unroll
      for (int j = 0; j < 8; ++j) { p2f[kf][j] = lo[j]; p2f[kf][j + 8] = hi[j]; }
    }
    // ---- PV: share V fragments between the two branches (8+8 WMMAs)
    #pragma unroll
    for (int kf = 0; kf < 2; ++kf)
      #pragma unroll
      for (int nt = 0; nt < 4; ++nt) {
        int vd = nt * 16 + l16, kb = kf * 32 + lh * 8;
        v8h lo = *(const v8h*)(Vt + vd * 64 + kb);
        v8h hi = *(const v8h*)(Vt + vd * 64 + kb + 16);
        v16h vf;
        #pragma unroll
        for (int j = 0; j < 8; ++j) { vf[j] = lo[j]; vf[j + 8] = hi[j]; }
        o1[nt] = __builtin_amdgcn_wmma_f32_16x16x32_f16(false, p1f[kf], false, vf, (short)0, o1[nt], false, false);
        o2[nt] = __builtin_amdgcn_wmma_f32_16x16x32_f16(false, p2f[kf], false, vf, (short)0, o2[nt], false, false);
      }
  }

  // ---- epilogue: 1/l, differential combine, RMS norm, scale, store
  #pragma unroll
  for (int i = 0; i < 8; ++i) {
    s1[i] = dpp_sum16(s1[i]);
    s2[i] = dpp_sum16(s2[i]);
  }
  float val[4][8], ss[8];
  #pragma unroll
  for (int i = 0; i < 8; ++i) ss[i] = 0.f;
  #pragma unroll
  for (int nt = 0; nt < 4; ++nt)
    #pragma unroll
    for (int i = 0; i < 8; ++i) {
      float v = o1[nt][i] / s1[i] - lam * (o2[nt][i] / s2[i]);
      val[nt][i] = v;
      ss[i] += v * v;
    }
  float rnorm[8];
  #pragma unroll
  for (int i = 0; i < 8; ++i) {
    ss[i] = dpp_sum16(ss[i]);
    rnorm[i] = rsqrtf(ss[i] * (1.0f / 64.0f) + 1e-5f);
  }

  #pragma unroll
  for (int nt = 0; nt < 4; ++nt) {
    float w = subw[nt * 16 + l16] * ONE_MINUS_LI;
    #pragma unroll
    for (int i = 0; i < 8; ++i) {
      int q = qw + 8 * lh + i;
      outh[((size_t)(b * SEQ_N + q)) * NDIM + h * 64 + nt * 16 + l16] =
          (_Float16)(val[nt][i] * rnorm[i] * w);
    }
  }
}

// ---------------------------------------------------------------------------
// Host launcher
// ---------------------------------------------------------------------------
extern "C" void kernel_launch(void* const* d_in, const int* in_sizes, int n_in,
                              void* d_out, int out_size, void* d_ws, size_t ws_size,
                              hipStream_t stream) {
  const float* x     = (const float*)d_in[0];
  const float* Wqkv  = (const float*)d_in[1];
  const float* Wproj = (const float*)d_in[2];
  const float* bproj = (const float*)d_in[3];
  const float* lq1   = (const float*)d_in[4];
  const float* lk1   = (const float*)d_in[5];
  const float* lq2   = (const float*)d_in[6];
  const float* lk2   = (const float*)d_in[7];
  const float* subw  = (const float*)d_in[8];
  float* out         = (float*)d_out;

  const size_t M = 2 * SEQ_N;            // 4096 rows
  char* ws = (char*)d_ws;
  _Float16* x_h     = (_Float16*)ws;                 ws += M * NDIM * 2;            // 8 MB
  _Float16* wqkv_h  = (_Float16*)ws;                 ws += (size_t)NDIM * 3072 * 2; // 6 MB
  _Float16* wproj_h = (_Float16*)ws;                 ws += (size_t)NDIM * NDIM * 2; // 2 MB
  _Float16* qkv_h   = (_Float16*)ws;                 ws += M * 3072 * 2;            // 24 MB
  _Float16* o_h     = (_Float16*)ws;                 ws += M * NDIM * 2;            // 8 MB
  float* lam        = (float*)ws;

  // 1) casts
  cast_f32_to_f16<<<(M * NDIM / 4 + 255) / 256, 256, 0, stream>>>(x, x_h, (int)(M * NDIM / 4));
  cast_f32_to_f16<<<(NDIM * 3072 / 4 + 255) / 256, 256, 0, stream>>>(Wqkv, wqkv_h, NDIM * 3072 / 4);
  cast_f32_to_f16<<<(NDIM * NDIM / 4 + 255) / 256, 256, 0, stream>>>(Wproj, wproj_h, NDIM * NDIM / 4);
  // 2) lambda scalar
  lambda_kernel<<<1, 32, 0, stream>>>(lq1, lk1, lq2, lk2, lam);
  // 3) QKV GEMM: (4096x1024) @ (1024x3072) -> f16
  wmma_gemm<true><<<dim3(3072 / 128, M / 128), 256, 0, stream>>>(
      x_h, wqkv_h, qkv_h, nullptr, nullptr, (int)M, 3072, NDIM);
  // 4) differential flash attention -> f16 o buffer
  diff_attn_kernel<<<dim3(SEQ_N / 128, 16, 2), 256, 0, stream>>>(qkv_h, lam, subw, o_h);
  // 5) projection GEMM + bias -> f32 out
  wmma_gemm<false><<<dim3(NDIM / 128, M / 128), 256, 0, stream>>>(
      o_h, wproj_h, nullptr, out, bproj, (int)M, NDIM, NDIM);
}
